// NonLocalAttention_23192823399127
// MI455X (gfx1250) — matrix operand order repaired and checked by reading, not compiled, not run on hardware
//
#include <hip/hip_runtime.h>

typedef unsigned short u16;
typedef __attribute__((ext_vector_type(16))) __bf16        v16bf;
typedef __attribute__((ext_vector_type(8)))  float         v8f;
typedef __attribute__((ext_vector_type(4)))  unsigned int  v4u;
typedef __attribute__((ext_vector_type(4)))  float         v4f;

#define N_   2
#define C_   256
#define CE_  128
#define HW_  6400
#define EPITCH 264   // padded bf16 row pitch for the embed LDS tile

union V16 {
  v16bf v;
  v4u   q[2];
  u16   us[16];
};

__device__ __forceinline__ u16 f2bf(float f) {
  union { float f; unsigned u; } x; x.f = f;
  unsigned r = x.u + 0x7FFFu + ((x.u >> 16) & 1u);  // round-to-nearest-even
  return (u16)(r >> 16);
}

__device__ __forceinline__ v8f wmma_bf16(const V16& a, const V16& b, v8f c) {
  // 8 args: (neg_a, A, neg_b, B, c_mod, C, reuse_a, reuse_b)
  return __builtin_amdgcn_wmma_f32_16x16x32_bf16(false, a.v, false, b.v,
                                                 (short)0, c, false, false);
}

// ---------------------------------------------------------------------------
// Kernel 1: fused 1x1 convs (three GEMMs over K=C=256) + bias + PReLU.
// Produces bf16 workspace panels:
//   qws  : (N, HW, Ce)  = PReLU(w1 @ x)^T   (A-operand layout for S GEMM)
//   ktws : (N, HW, Ce)  = PReLU(w2 @ x)^T   (B-operand columns contiguous)
//   vvws : (N, C,  HW)  = PReLU(wa @ x)     (B-operand for P@V contiguous)
// One block per (n, 64-wide i tile); input tile staged transposed in LDS.
// ---------------------------------------------------------------------------
__global__ __launch_bounds__(256)
void embed_kernel(const float* __restrict__ inp,
                  const float* __restrict__ w1, const float* __restrict__ b1,
                  const float* __restrict__ a1,
                  const float* __restrict__ w2, const float* __restrict__ b2,
                  const float* __restrict__ a2,
                  const float* __restrict__ wa, const float* __restrict__ ba,
                  const float* __restrict__ aa,
                  u16* __restrict__ qws, u16* __restrict__ ktws,
                  u16* __restrict__ vvws) {
  __shared__ u16 tile[64 * EPITCH];            // [i][c] transposed, bf16

  const int blk = blockIdx.x;                  // N * HW/64 blocks
  const int n   = blk / (HW_ / 64);
  const int i0  = (blk % (HW_ / 64)) * 64;
  const int t   = threadIdx.x;

  // Stage input[n][c][i0..i0+63] -> tile[i][c] as bf16 (coalesced b128 reads)
  const float* ibase = inp + (size_t)n * C_ * HW_ + i0;
  for (int s = 0; s < 16; ++s) {
    int qd = t + 256 * s;                      // 4096 float4 quads
    int c  = qd >> 4;
    int i4 = (qd & 15) * 4;
    v4f v = *(const v4f*)(ibase + (size_t)c * HW_ + i4);
    tile[(i4 + 0) * EPITCH + c] = f2bf(v.x);
    tile[(i4 + 1) * EPITCH + c] = f2bf(v.y);
    tile[(i4 + 2) * EPITCH + c] = f2bf(v.z);
    tile[(i4 + 3) * EPITCH + c] = f2bf(v.w);
  }
  __syncthreads();

  const int wv = t >> 5, l = t & 31;
  const int ln = l & 15, hf = l >> 4;

  // 512 output rows (Ce + Ce + C) = 32 row tiles of 16, round-robin by wave
  for (int rt = wv; rt < 32; rt += 8) {
    const float *W, *Bb; float alpha; int Rl;
    u16* OP; int sI, sO;                       // branchless store descriptor
    if (rt < 8)       { W = w1; Bb = b1; alpha = a1[0]; Rl = rt * 16;
                        OP = qws  + (size_t)n * HW_ * CE_; sI = CE_; sO = 1; }
    else if (rt < 16) { W = w2; Bb = b2; alpha = a2[0]; Rl = (rt - 8) * 16;
                        OP = ktws + (size_t)n * HW_ * CE_; sI = CE_; sO = 1; }
    else              { W = wa; Bb = ba; alpha = aa[0]; Rl = (rt - 16) * 16;
                        OP = vvws + (size_t)n * C_ * HW_;  sI = 1;   sO = HW_; }

    // Preload A (weights) for this row tile: 8 chunks of K=32, bf16 A-layout
    V16 A[8];
#pragma unroll
    for (int kc = 0; kc < 8; ++kc) {
      const float* wr = W + (size_t)(Rl + ln) * C_ + kc * 32 + hf * 8;
#pragma unroll
      for (int e = 0; e < 8; ++e) {
        A[kc].us[e]     = f2bf(wr[e]);        // K = base + 0..7
        A[kc].us[e + 8] = f2bf(wr[e + 16]);   // K = base + 16..23
      }
    }

    for (int it = 0; it < 4; ++it) {
      v8f acc = {0, 0, 0, 0, 0, 0, 0, 0};
#pragma unroll
      for (int kc = 0; kc < 8; ++kc) {
        V16 B;
        const u16* br = &tile[(it * 16 + ln) * EPITCH + kc * 32 + hf * 8];
        B.q[0] = *(const v4u*)br;             // K run 0..7  (ds_load_b128)
        B.q[1] = *(const v4u*)(br + 16);      // K run 16..23
        acc = wmma_bf16(A[kc], B, acc);
      }
      int icol = i0 + it * 16 + ln;           // D-layout: lane%16 = N (i col)
#pragma unroll
      for (int r = 0; r < 8; ++r) {
        int o = Rl + r + 8 * hf;              // D-layout: VGPR r -> row
        float v = acc[r] + Bb[o];
        v = (v >= 0.f) ? v : alpha * v;       // PReLU (single shared alpha)
        OP[(size_t)icol * sI + (size_t)o * sO] = f2bf(v);
      }
    }
  }
}

// ---------------------------------------------------------------------------
// Kernel 2: fused flash-style attention. Each wave owns a 16-row tile of the
// 6400x6400 score matrix and streams 64-column steps:
//   S (16x64) = Q_tile @ K_tile        -> 16 bf16 WMMAs (4 indep chains)
//   online softmax in fp32 on D-layout (half-wave shuffle reductions)
//   P -> per-wave LDS -> A-layout      (same-wave DS ops are in-order)
//   O (16x256) += P @ V                -> 32 bf16 WMMAs
// Final: out = O / rowsum + input. K/V panels are L2-resident (~5MB/batch),
// next tile prefetched with global_prefetch while WMMAs run.
// ---------------------------------------------------------------------------
__global__ __launch_bounds__(256)
void attn_kernel(const float* __restrict__ inp,
                 const u16* __restrict__ qws, const u16* __restrict__ ktws,
                 const u16* __restrict__ vvws, float* __restrict__ out) {
  __shared__ u16 plds[8 * 16 * 64];            // 2KB per wave: P tile 16x64

  const int blk = blockIdx.x;                  // N * HW/128 blocks
  const int n   = blk / (HW_ / 128);
  const int rb  = blk % (HW_ / 128);
  const int t = threadIdx.x, wv = t >> 5, l = t & 31;
  const int ln = l & 15, hf = l >> 4;
  const int i0 = rb * 128 + wv * 16;

  // Preload A_q: rows i0..i0+15 over Ce=128 -> 4 K=32 chunks (A-layout)
  V16 Aq[4];
  {
    const u16* qr = qws + ((size_t)n * HW_ + i0 + ln) * CE_ + hf * 8;
#pragma unroll
    for (int kc = 0; kc < 4; ++kc) {
      Aq[kc].q[0] = *(const v4u*)(qr + kc * 32);
      Aq[kc].q[1] = *(const v4u*)(qr + kc * 32 + 16);
    }
  }

  v8f O[16];
  const v8f zer = {0, 0, 0, 0, 0, 0, 0, 0};
#pragma unroll
  for (int ct = 0; ct < 16; ++ct) O[ct] = zer;
  float mrow[8], lrow[8];
#pragma unroll
  for (int r = 0; r < 8; ++r) { mrow[r] = -1e30f; lrow[r] = 0.f; }

  u16* pw = &plds[wv * 16 * 64];
  const u16* ktb = ktws + (size_t)n * HW_ * CE_;
  const u16* vvb = vvws + (size_t)n * C_ * HW_;

  for (int j0 = 0; j0 < HW_; j0 += 64) {
    // Prefetch next column tile of K and V into L2/L0 (global_prefetch_b8)
    if (j0 + 64 < HW_) {
      __builtin_prefetch(ktb + (size_t)(j0 + 64 + l) * CE_, 0, 1);
      __builtin_prefetch(ktb + (size_t)(j0 + 96 + l) * CE_, 0, 1);
#pragma unroll
      for (int pc = 0; pc < 8; ++pc)
        __builtin_prefetch(vvb + (size_t)(pc * 32 + l) * HW_ + j0 + 64, 0, 1);
    }

    // S tiles: four 16x16 logit blocks, K=128 reduction, 4 indep WMMA chains
    v8f S[4] = {zer, zer, zer, zer};
#pragma unroll
    for (int kc = 0; kc < 4; ++kc) {
      const u16* k0 = ktb + (size_t)(j0 + ln) * CE_ + kc * 32 + hf * 8;
#pragma unroll
      for (int st = 0; st < 4; ++st) {
        V16 B;
        const u16* kp = k0 + (size_t)(st * 16) * CE_;
        B.q[0] = *(const v4u*)kp;
        B.q[1] = *(const v4u*)(kp + 16);
        S[st] = wmma_bf16(Aq[kc], B, S[st]);
      }
    }

    // Online softmax: D-layout row m = r + 8*hf lives across lanes of a half
#pragma unroll
    for (int r = 0; r < 8; ++r) {
      float tm = fmaxf(fmaxf(S[0][r], S[1][r]), fmaxf(S[2][r], S[3][r]));
      tm = fmaxf(tm, __shfl_xor(tm, 1));
      tm = fmaxf(tm, __shfl_xor(tm, 2));
      tm = fmaxf(tm, __shfl_xor(tm, 4));
      tm = fmaxf(tm, __shfl_xor(tm, 8));
      float mn = fmaxf(mrow[r], tm);
      float sc = __expf(mrow[r] - mn);
      mrow[r] = mn;
      float rs = 0.f;
#pragma unroll
      for (int st = 0; st < 4; ++st) {
        float p = __expf(S[st][r] - mn);
        S[st][r] = p;
        rs += p;
      }
      rs += __shfl_xor(rs, 1);
      rs += __shfl_xor(rs, 2);
      rs += __shfl_xor(rs, 4);
      rs += __shfl_xor(rs, 8);
      lrow[r] = lrow[r] * sc + rs;
#pragma unroll
      for (int ct = 0; ct < 16; ++ct) O[ct][r] *= sc;   // rescale accumulator
    }

    // P (D-layout) -> LDS row-major 16x64 bf16; same-wave DS ops are in-order
#pragma unroll
    for (int r = 0; r < 8; ++r) {
      int mi = r + 8 * hf;
#pragma unroll
      for (int st = 0; st < 4; ++st)
        pw[mi * 64 + st * 16 + ln] = f2bf(S[st][r]);
    }
    // Reload P in A-layout: two K=32 chunks; lane ln = row, K split hf*8/+16
    V16 P0, P1;
    {
      const u16* pr = pw + ln * 64 + hf * 8;
      P0.q[0] = *(const v4u*)pr;
      P0.q[1] = *(const v4u*)(pr + 16);
      P1.q[0] = *(const v4u*)(pr + 32);
      P1.q[1] = *(const v4u*)(pr + 48);
    }
    // O += P @ V : 16 column tiles of C=256, V rows contiguous in vvws
#pragma unroll
    for (int ct = 0; ct < 16; ++ct) {
      V16 Bv0, Bv1;
      const u16* vr = vvb + (size_t)(ct * 16 + ln) * HW_ + j0 + hf * 8;
      Bv0.q[0] = *(const v4u*)vr;
      Bv0.q[1] = *(const v4u*)(vr + 16);
      Bv1.q[0] = *(const v4u*)(vr + 32);
      Bv1.q[1] = *(const v4u*)(vr + 48);
      O[ct] = wmma_bf16(P0, Bv0, O[ct]);
      O[ct] = wmma_bf16(P1, Bv1, O[ct]);
    }
  }

  // Epilogue: normalize, add residual, store to NCHW
#pragma unroll
  for (int r = 0; r < 8; ++r) lrow[r] = 1.0f / lrow[r];
#pragma unroll
  for (int ct = 0; ct < 16; ++ct) {
    int c = ct * 16 + ln;
#pragma unroll
    for (int r = 0; r < 8; ++r) {
      int i = i0 + r + 8 * hf;
      size_t idx = ((size_t)n * C_ + c) * HW_ + i;
      out[idx] = O[ct][r] * lrow[r] + inp[idx];
    }
  }
}

extern "C" void kernel_launch(void* const* d_in, const int* in_sizes, int n_in,
                              void* d_out, int out_size, void* d_ws, size_t ws_size,
                              hipStream_t stream) {
  (void)in_sizes; (void)n_in; (void)out_size; (void)ws_size;
  const float* inp = (const float*)d_in[0];
  const float* w1  = (const float*)d_in[1];
  const float* b1  = (const float*)d_in[2];
  const float* a1  = (const float*)d_in[3];
  const float* w2  = (const float*)d_in[4];
  const float* b2  = (const float*)d_in[5];
  const float* a2  = (const float*)d_in[6];
  const float* wa  = (const float*)d_in[7];
  const float* ba  = (const float*)d_in[8];
  const float* aa  = (const float*)d_in[9];
  float* out = (float*)d_out;

  u16* qws  = (u16*)d_ws;                                  // (N,HW,Ce) bf16
  u16* ktws = qws  + (size_t)N_ * HW_ * CE_;               // (N,HW,Ce) bf16
  u16* vvws = ktws + (size_t)N_ * HW_ * CE_;               // (N,C,HW)  bf16

  embed_kernel<<<dim3(N_ * (HW_ / 64)), dim3(256), 0, stream>>>(
      inp, w1, b1, a1, w2, b2, a2, wa, ba, aa, qws, ktws, vvws);
  attn_kernel<<<dim3(N_ * (HW_ / 128)), dim3(256), 0, stream>>>(
      inp, qws, ktws, vvws, out);
}
